// VarianceAdaptor_17781164605702
// MI455X (gfx1250) — compile-verified, matrix operand order checked by reading, not run on hardware
//
#include <hip/hip_runtime.h>

typedef __attribute__((ext_vector_type(16))) __bf16 v16bf;
typedef __attribute__((ext_vector_type(8)))  float  v8f;

#define Bsz   32
#define Ssz   1024
#define Hsz   512
#define Fsz   512
#define MLsz  4096
#define KSTEPS 48          // (3*512)/32 K-steps of depth 32
#define XPITCH 520         // ushort pitch for A slice rows (512 + 8 pad, 16B-aligned)

// ---------- bf16 helpers (RNE, bit-level; no scalar __bf16 math) ----------
__device__ __forceinline__ unsigned f2bfbits(float f) {
  union { float f; unsigned u; } c; c.f = f;
  unsigned u = c.u;
  return ((u + 0x7FFFu + ((u >> 16) & 1u)) >> 16) & 0xFFFFu;
}

// ---------- weight prepack into WMMA B-fragment layout ----------
// Fragment order: [t (48 K-steps)][jj (32 N-tiles)][lane (32)][8 packed K-pairs]
// Lane<16 holds column n=lane, K pairs {0..7},{16..23}; lane>=16 column n=lane-16,
// K pairs {8..15},{24..31} — mirrors the 16-bit B 32x16 VGPR layout.
__global__ __launch_bounds__(256) void pack_w(const float* __restrict__ w,
                                              unsigned int* __restrict__ wp) {
  int idx = blockIdx.x * 256 + threadIdx.x;       // 48*32*32*8 = 393216 total
  if (idx >= 48 * 32 * 32 * 8) return;
  int v = idx & 7;
  int L = (idx >> 3) & 31;
  int j = (idx >> 8) & 31;
  int t = idx >> 13;
  int n   = (j << 4) + (L & 15);
  int kk0 = ((L < 16) ? 0 : 8) + ((v < 4) ? (2 * v) : (16 + 2 * (v - 4)));
  int k   = t >> 4;                 // conv tap 0..2
  int h0  = (t & 15) << 5;          // input-channel base
  float lo = w[((size_t)(k * Hsz + h0 + kk0)) * Fsz + n];
  float hi = w[((size_t)(k * Hsz + h0 + kk0 + 1)) * Fsz + n];
  wp[idx] = f2bfbits(lo) | (f2bfbits(hi) << 16);
}

// ---------- fused conv1d(K=3) + bias + relu + layernorm (+ optional lin dot) ----------
template <bool FIRST>
__global__ __launch_bounds__(256) void conv_ln(
    const float*          __restrict__ xin,     // FIRST: x f32
    const unsigned short* __restrict__ ain,     // !FIRST: h1 bf16
    const unsigned int*   __restrict__ wp,      // prepacked weights
    const float* __restrict__ bias,
    const float* __restrict__ gamma, const float* __restrict__ beta,
    unsigned short* __restrict__ hout,          // FIRST: h1 bf16 out
    const float* __restrict__ lin_w, const float* __restrict__ lin_b,
    const int*   __restrict__ src_lens,
    float*       __restrict__ logdur)           // !FIRST out
{
  __shared__ __align__(16) char smem[65536];
  unsigned short* xs = (unsigned short*)smem;   // phase 1: [34][XPITCH] bf16
  float*          os = (float*)smem;            // phase 2: [32][512]   f32

  const int tid  = threadIdx.x;
  const int wave = tid >> 5;
  const int lane = tid & 31;
  const int bb   = blockIdx.x >> 5;             // batch
  const int s0   = (blockIdx.x & 31) << 5;      // token tile base

  // Stage A slice (rows s0-1 .. s0+32, zero-padded) into LDS as bf16.
  if (FIRST) {
    for (int i = tid; i < 34 * 128; i += 256) {
      int r = i >> 7, c4 = (i & 127) << 2;
      int s = s0 + r - 1;
      float4 v = make_float4(0.f, 0.f, 0.f, 0.f);
      if (s >= 0 && s < Ssz)
        v = *(const float4*)(xin + ((size_t)(bb * Ssz + s)) * Hsz + c4);
      uint2 pk;
      pk.x = f2bfbits(v.x) | (f2bfbits(v.y) << 16);
      pk.y = f2bfbits(v.z) | (f2bfbits(v.w) << 16);
      *(uint2*)(xs + r * XPITCH + c4) = pk;
    }
  } else {
    for (int i = tid; i < 34 * 64; i += 256) {
      int r = i >> 6, c8 = (i & 63) << 3;
      int s = s0 + r - 1;
      uint4 v = make_uint4(0u, 0u, 0u, 0u);
      if (s >= 0 && s < Ssz)
        v = *(const uint4*)(ain + ((size_t)(bb * Ssz + s)) * Fsz + c8);
      *(uint4*)(xs + r * XPITCH + c8) = v;
    }
  }
  __syncthreads();

  v8f acc[2][4];
#pragma unroll
  for (int i = 0; i < 2; ++i)
#pragma unroll
    for (int j = 0; j < 4; ++j)
#pragma unroll
      for (int e = 0; e < 8; ++e) acc[i][j][e] = 0.f;

  const int lrow = lane & 15;
  const int sel  = (lane < 16) ? 0 : 8;   // A-matrix lane-half K offset

  union FRAG { v16bf v; uint4 u[2]; };

  for (int t = 0; t < KSTEPS; ++t) {
    const int k  = t >> 4;                // conv tap -> row shift in LDS slice
    const int h0 = (t & 15) << 5;
    FRAG a[2];
#pragma unroll
    for (int mi = 0; mi < 2; ++mi) {
      const unsigned short* ap = xs + (mi * 16 + lrow + k) * XPITCH + h0 + sel;
      a[mi].u[0] = *(const uint4*)ap;          // K {0..7} or {8..15}
      a[mi].u[1] = *(const uint4*)(ap + 16);   // K {16..23} or {24..31}
    }
#pragma unroll
    for (int j = 0; j < 4; ++j) {
      const int jj = (wave << 2) + j;
      FRAG bf;
      const uint4* bp = (const uint4*)(wp + ((size_t)(t * 32 + jj) * 32 + lane) * 8);
      bf.u[0] = bp[0];
      bf.u[1] = bp[1];
#pragma unroll
      for (int mi = 0; mi < 2; ++mi)
        acc[mi][j] = __builtin_amdgcn_wmma_f32_16x16x32_bf16(
            false, a[mi].v, false, bf.v, (short)0, acc[mi][j], false, false);
    }
  }
  __syncthreads();   // xs dead; reuse smem as the f32 output tile

  // Epilogue: bias + relu into LDS tile (C layout: lane half selects M+8).
#pragma unroll
  for (int mi = 0; mi < 2; ++mi) {
    const int rbase = mi * 16 + ((lane < 16) ? 0 : 8);
#pragma unroll
    for (int j = 0; j < 4; ++j) {
      const int col = (wave << 6) + (j << 4) + (lane & 15);
      const float bv = bias[col];
#pragma unroll
      for (int v = 0; v < 8; ++v)
        os[(rbase + v) * Fsz + col] = fmaxf(acc[mi][j][v] + bv, 0.f);
    }
  }
  __syncthreads();

  // LayerNorm over F=512 per row; wave w owns rows 4w..4w+3 (wave32 shfl reduce).
  for (int rr = 0; rr < 4; ++rr) {
    const int row   = (wave << 2) + rr;
    const int cbase = lane << 4;
    float vals[16];
    float sum = 0.f, sq = 0.f;
#pragma unroll
    for (int q = 0; q < 4; ++q) {
      float4 v4 = *(const float4*)(os + row * Fsz + cbase + (q << 2));
      vals[q * 4 + 0] = v4.x; vals[q * 4 + 1] = v4.y;
      vals[q * 4 + 2] = v4.z; vals[q * 4 + 3] = v4.w;
      sum += v4.x + v4.y + v4.z + v4.w;
      sq  += v4.x * v4.x + v4.y * v4.y + v4.z * v4.z + v4.w * v4.w;
    }
#pragma unroll
    for (int m = 16; m > 0; m >>= 1) {
      sum += __shfl_xor(sum, m, 32);
      sq  += __shfl_xor(sq,  m, 32);
    }
    const float mean = sum * (1.f / 512.f);
    const float inv  = rsqrtf(sq * (1.f / 512.f) - mean * mean + 1e-5f);
#pragma unroll
    for (int q = 0; q < 4; ++q) {
      float4 g4 = *(const float4*)(gamma + cbase + (q << 2));
      float4 b4 = *(const float4*)(beta  + cbase + (q << 2));
      vals[q * 4 + 0] = (vals[q * 4 + 0] - mean) * inv * g4.x + b4.x;
      vals[q * 4 + 1] = (vals[q * 4 + 1] - mean) * inv * g4.y + b4.y;
      vals[q * 4 + 2] = (vals[q * 4 + 2] - mean) * inv * g4.z + b4.z;
      vals[q * 4 + 3] = (vals[q * 4 + 3] - mean) * inv * g4.w + b4.w;
    }
    if constexpr (FIRST) {
      union { uint4 u[2]; unsigned short h[16]; } ob;
#pragma unroll
      for (int e = 0; e < 16; ++e) ob.h[e] = (unsigned short)f2bfbits(vals[e]);
      uint4* dst = (uint4*)(hout + ((size_t)(bb * Ssz + s0 + row)) * Fsz + cbase);
      dst[0] = ob.u[0];
      dst[1] = ob.u[1];
    } else {
      float ld = 0.f;
#pragma unroll
      for (int q = 0; q < 4; ++q) {
        float4 lw = *(const float4*)(lin_w + cbase + (q << 2));
        ld += vals[q * 4 + 0] * lw.x + vals[q * 4 + 1] * lw.y +
              vals[q * 4 + 2] * lw.z + vals[q * 4 + 3] * lw.w;
      }
#pragma unroll
      for (int m = 16; m > 0; m >>= 1) ld += __shfl_xor(ld, m, 32);
      if (lane == 0) {
        const int sg = s0 + row;
        logdur[bb * Ssz + sg] = (sg < src_lens[bb]) ? (ld + lin_b[0]) : 0.f;
      }
    }
  }
}

// ---------- per-batch masked cumsum of duration ----------
__global__ __launch_bounds__(1024) void dur_scan(
    const int* __restrict__ duration, const int* __restrict__ src_lens,
    int* __restrict__ csum, float* __restrict__ out_dur, float* __restrict__ out_tlen) {
  __shared__ int buf[1024];
  const int b = blockIdx.x, t = threadIdx.x;
  const int v = (t < src_lens[b]) ? duration[b * Ssz + t] : 0;
  out_dur[b * Ssz + t] = (float)v;
  buf[t] = v;
  __syncthreads();
  for (int off = 1; off < 1024; off <<= 1) {
    const int add = (t >= off) ? buf[t - off] : 0;
    __syncthreads();
    buf[t] += add;
    __syncthreads();
  }
  csum[b * Ssz + t] = buf[t];
  if (t == 1023) out_tlen[b] = (float)buf[t];
}

// ---------- length-regulated expansion (searchsorted + gather) ----------
__global__ __launch_bounds__(256) void expand_k(
    const float* __restrict__ x, const int* __restrict__ csum,
    float* __restrict__ out_exp, float* __restrict__ out_mask) {
  __shared__ int cs[1024];
  __shared__ int idxs[64];
  const int b  = blockIdx.y;
  const int p0 = blockIdx.x << 6;
  const int t  = threadIdx.x;
  for (int i = t; i < 1024; i += 256) cs[i] = csum[b * Ssz + i];
  __syncthreads();
  const int tlen = cs[1023];
  if (t < 64) {
    const int p = p0 + t;
    int lo = 0, hi = 1024;                 // upper_bound: first csum[i] > p
    while (lo < hi) {
      const int mid = (lo + hi) >> 1;
      if (cs[mid] <= p) lo = mid + 1; else hi = mid;
    }
    idxs[t] = min(lo, Ssz - 1);
    out_mask[b * MLsz + p] = (p < tlen) ? 0.f : 1.f;
  }
  __syncthreads();
  const int  pl    = t >> 2, q = t & 3;
  const int  p     = p0 + pl;
  const int  idx   = idxs[pl];
  const bool valid = p < tlen;
  const float4* src = (const float4*)(x + ((size_t)b * Ssz + idx) * Hsz + q * 128);
  float4*       dst = (float4*)(out_exp + ((size_t)b * MLsz + p) * Hsz + q * 128);
  const float4  z   = make_float4(0.f, 0.f, 0.f, 0.f);
#pragma unroll
  for (int r = 0; r < 32; ++r) dst[r] = valid ? src[r] : z;
}

extern "C" void kernel_launch(void* const* d_in, const int* in_sizes, int n_in,
                              void* d_out, int out_size, void* d_ws, size_t ws_size,
                              hipStream_t stream) {
  const float* x        = (const float*)d_in[0];
  const int*   duration = (const int*)  d_in[1];
  const int*   src_lens = (const int*)  d_in[2];
  // d_in[3] = src_mask: recomputed from src_lens instead (dtype-safe)
  const float* conv1_w  = (const float*)d_in[4];
  const float* conv1_b  = (const float*)d_in[5];
  const float* ln1_g    = (const float*)d_in[6];
  const float* ln1_b    = (const float*)d_in[7];
  const float* conv2_w  = (const float*)d_in[8];
  const float* conv2_b  = (const float*)d_in[9];
  const float* ln2_g    = (const float*)d_in[10];
  const float* ln2_b    = (const float*)d_in[11];
  const float* lin_w    = (const float*)d_in[12];
  const float* lin_b    = (const float*)d_in[13];

  char* ws = (char*)d_ws;
  unsigned int*   w1p  = (unsigned int*)(ws);                                // 1.5 MB
  unsigned int*   w2p  = (unsigned int*)(ws + (size_t)(2u << 20));           // 1.5 MB
  unsigned short* h1b  = (unsigned short*)(ws + (size_t)(4u << 20));         // 32  MB
  int*            csum = (int*)(ws + (size_t)(4u << 20) + (size_t)(32u << 20));

  float* out        = (float*)d_out;
  float* out_exp    = out;                                   // 32*4096*512
  float* out_logdur = out + (size_t)67108864;                // 32*1024
  float* out_dur    = out_logdur + 32768;                    // 32*1024
  float* out_tlen   = out_dur + 32768;                       // 32
  float* out_mask   = out_tlen + 32;                         // 32*4096

  pack_w<<<1536, 256, 0, stream>>>(conv1_w, w1p);
  pack_w<<<1536, 256, 0, stream>>>(conv2_w, w2p);

  conv_ln<true><<<1024, 256, 0, stream>>>(
      x, nullptr, w1p, conv1_b, ln1_g, ln1_b, h1b,
      nullptr, nullptr, nullptr, nullptr);

  conv_ln<false><<<1024, 256, 0, stream>>>(
      nullptr, h1b, w2p, conv2_b, ln2_g, ln2_b, nullptr,
      lin_w, lin_b, src_lens, out_logdur);

  dur_scan<<<32, 1024, 0, stream>>>(duration, src_lens, csum, out_dur, out_tlen);

  expand_k<<<dim3(64, 32), 256, 0, stream>>>(x, csum, out_exp, out_mask);
}
